// MLA_51565377356265
// MI455X (gfx1250) — matrix operand order repaired and checked by reading, not compiled, not run on hardware
//
#include <hip/hip_runtime.h>
#include <hip/hip_fp16.h>

typedef __attribute__((ext_vector_type(16))) _Float16 v16h;
typedef __attribute__((ext_vector_type(8)))  float    v8f;

#define D_MODEL 4096
#define N_HEADS 32
#define D_H     128
#define QK_NOPE 64
#define QK_ROPE 64
#define PROJ    2048
#define SEQ     2048

__device__ __forceinline__ v8f wmma_f16(v16h a, v16h b, v8f c) {
    // D = A(16x32 f16) * B(32x16 f16) + C(16x16 f32)
    return __builtin_amdgcn_wmma_f32_16x16x32_f16(
        /*neg_a=*/false, a, /*neg_b=*/false, b,
        /*c_mod=*/(short)0, c, /*reuse_a=*/false, /*reuse_b=*/false);
}

// ---------------------------------------------------------------------------
// GEMM: C[M,N] = A[M,K] @ B[K,N] (+bias), fp32 row-major, f16 WMMA inside.
// 256 threads = 8 waves. Block tile 128x128, BK=32.
// Wave (wm,wn) in a 4x2 grid computes a 32x64 sub-tile: 2x4 WMMA accumulators,
// 8 WMMAs per K-step vs 6 fragment loads (all ds_load_b128-contiguous).
// B is staged TRANSPOSED in LDS so B-fragments read contiguous k-runs.
// N-edge guarded (handles the N=64 k_rope projection). M,K multiples of 128/32.
// ---------------------------------------------------------------------------
template<bool HAS_BIAS>
__global__ void __launch_bounds__(256)
gemm_wmma_f16(const float* __restrict__ A, const float* __restrict__ B,
              const float* __restrict__ bias, float* __restrict__ C,
              int M, int N, int K)
{
    __shared__ _Float16 As [128][32 + 8];   // [m][k], row stride 80B (16B mult)
    __shared__ _Float16 Bst[128][32 + 8];   // [n][k] transposed

    const int tid  = threadIdx.x;
    const int w    = tid >> 5;
    const int lane = tid & 31;
    const int g    = lane >> 4;
    const int ln   = lane & 15;
    const int wm   = w >> 1;                // 0..3 -> 32-row strip
    const int wn   = w & 1;                 // 0..1 -> 64-col strip
    const int m0   = blockIdx.y * 128;
    const int n0   = blockIdx.x * 128;

    v8f acc[2][4] = {};

    for (int k0 = 0; k0 < K; k0 += 32) {
        // Stage A tile (128x32) and B tile (32x128, transposed), f32 -> f16.
        #pragma unroll
        for (int r = 0; r < 16; ++r) {
            int i  = tid + 256 * r;
            int ar = i >> 5, ac = i & 31;
            As[ar][ac] = (_Float16)A[(size_t)(m0 + ar) * K + k0 + ac];
        }
        #pragma unroll
        for (int r = 0; r < 16; ++r) {
            int i   = tid + 256 * r;
            int br  = i >> 7, bc = i & 127;
            int col = n0 + bc;
            float vv = (col < N) ? B[(size_t)(k0 + br) * N + col] : 0.0f;
            Bst[bc][br] = (_Float16)vv;
        }
        if (k0 + 32 < K) {  // hint next A tile into cache (global_prefetch_b8)
            __builtin_prefetch(&A[(size_t)(m0 + (tid >> 1)) * K + k0 + 32 +
                                  (tid & 1) * 16], 0, 0);
        }
        __syncthreads();

        // A fragments (2 per wave): 16 contiguous f16 in two 16B runs.
        v16h a[2];
        #pragma unroll
        for (int mi = 0; mi < 2; ++mi)
            #pragma unroll
            for (int j = 0; j < 8; ++j) {
                int kb = g * 16 + 2 * j;
                a[mi][2 * j]     = As[wm * 32 + mi * 16 + ln][kb];
                a[mi][2 * j + 1] = As[wm * 32 + mi * 16 + ln][kb + 1];
            }
        #pragma unroll
        for (int ti = 0; ti < 4; ++ti) {
            // B fragment from transposed LDS: contiguous k per lane.
            v16h b;
            #pragma unroll
            for (int j = 0; j < 8; ++j) {
                int kk = g * 16 + 2 * j;
                b[2 * j]     = Bst[wn * 64 + ti * 16 + ln][kk];
                b[2 * j + 1] = Bst[wn * 64 + ti * 16 + ln][kk + 1];
            }
            #pragma unroll
            for (int mi = 0; mi < 2; ++mi)
                acc[mi][ti] = wmma_f16(a[mi], b, acc[mi][ti]);
        }
        __syncthreads();
    }

    #pragma unroll
    for (int mi = 0; mi < 2; ++mi)
        #pragma unroll
        for (int ti = 0; ti < 4; ++ti)
            #pragma unroll
            for (int j = 0; j < 8; ++j) {
                int row = m0 + wm * 32 + mi * 16 + g * 8 + j;
                int col = n0 + wn * 64 + ti * 16 + ln;
                if (col < N) {
                    float vv = acc[mi][ti][j];
                    if (HAS_BIAS) vv += bias[col];
                    C[(size_t)row * N + col] = vv;
                }
            }
}

// ---------------------------------------------------------------------------
// RoPE in place. grid = (SEQ, N_HEADS+1): y < N_HEADS rotates q head y's rope
// half; y == N_HEADS rotates k_rope. Read, barrier, write (rotate-half hazard).
// ---------------------------------------------------------------------------
__global__ void __launch_bounds__(64)
rope_kernel(float* __restrict__ q, float* __restrict__ kr,
            const int* __restrict__ past_len)
{
    const int s  = blockIdx.x;
    const int hh = blockIdx.y;
    const int i  = threadIdx.x;                 // 0..63
    const float pos  = (float)(past_len[0] + s);
    const int   fi   = i & 31;                  // sin/cos tiled (1,2)
    const float freq = __powf(10000.0f, -(float)(2 * fi) / 128.0f);
    float sn, cs;
    __sincosf(pos * freq, &sn, &cs);

    float* base = (hh < N_HEADS)
        ? (q + (size_t)s * D_MODEL + hh * D_H + QK_NOPE)
        : (kr + (size_t)s * QK_ROPE);

    float xv   = base[i];
    float part = (i < 32) ? -base[i + 32] : base[i - 32];
    float r    = xv * cs + part * sn;
    __syncthreads();
    base[i] = r;
}

// ---------------------------------------------------------------------------
// Flash attention (non-causal). grid = (SEQ/128, N_HEADS), 256 threads/8 waves.
// Block: one head x 128 query rows; wave w owns rows [w*16, w*16+16).
// Per 64-key tile: stage K=[k_nope|k_rope] row-major and V TRANSPOSED to LDS
// as f16, S = (Q/sqrt(d)) @ K^T via WMMA, online softmax, O += P @ V via WMMA.
// All WMMA fragment LDS reads are 16B-contiguous (ds_load_b128).
// ---------------------------------------------------------------------------
__global__ void __launch_bounds__(256)
attn_flash_wmma(const float* __restrict__ q,  const float* __restrict__ kn,
                const float* __restrict__ kr, const float* __restrict__ v,
                float* __restrict__ out)
{
    __shared__ _Float16 Ks [64][D_H + 8];      // [key][d]   row 272B
    __shared__ _Float16 Vst[D_H][64 + 8];      // [d][key]   row 144B
    __shared__ _Float16 Ps [8][16][64 + 8];    // per-wave P round-trip

    const int tid  = threadIdx.x;
    const int w    = tid >> 5;
    const int lane = tid & 31;
    const int g    = lane >> 4;
    const int ln   = lane & 15;
    const int h    = blockIdx.y;
    const int q0   = blockIdx.x * 128;
    const float scale = 0.088388347648318447f;  // 1/sqrt(128), folded into Q

    // Q A-fragments for this wave's 16 rows: d=128 -> 4 chunks of K=32.
    v16h aq[4];
    const float* qrow = q + (size_t)(q0 + w * 16 + ln) * D_MODEL + h * D_H;
    #pragma unroll
    for (int c = 0; c < 4; ++c)
        #pragma unroll
        for (int j = 0; j < 8; ++j) {
            int kb = 32 * c + g * 16 + 2 * j;
            aq[c][2 * j]     = (_Float16)(qrow[kb] * scale);
            aq[c][2 * j + 1] = (_Float16)(qrow[kb + 1] * scale);
        }

    v8f   oacc[8] = {};
    float mrow[8], lrow[8];
    #pragma unroll
    for (int j = 0; j < 8; ++j) { mrow[j] = -1e30f; lrow[j] = 0.0f; }

    for (int kt = 0; kt < SEQ / 64; ++kt) {
        // Stage K (k_nope head slice + shared k_rope) and V (transposed).
        #pragma unroll 4
        for (int r = 0; r < 32; ++r) {
            int i   = tid + 256 * r;
            int row = i >> 7, col = i & 127;
            int key = kt * 64 + row;
            float kvv = (col < QK_NOPE)
                ? kn[(size_t)key * PROJ + h * QK_NOPE + col]
                : kr[(size_t)key * QK_ROPE + (col - QK_NOPE)];
            Ks[row][col]  = (_Float16)kvv;
            Vst[col][row] = (_Float16)v[(size_t)key * D_MODEL + h * D_H + col];
        }
        __syncthreads();

        // Scores: S(16x64) = Qw @ Ktile^T ; B[k][n] = K[key n][k].
        v8f sacc[4];
        #pragma unroll
        for (int t = 0; t < 4; ++t) {
            v8f s = {};
            #pragma unroll
            for (int c = 0; c < 4; ++c) {
                v16h b;
                #pragma unroll
                for (int j = 0; j < 8; ++j) {
                    int kk = 32 * c + g * 16 + 2 * j;
                    b[2 * j]     = Ks[t * 16 + ln][kk];
                    b[2 * j + 1] = Ks[t * 16 + ln][kk + 1];
                }
                s = wmma_f16(aq[c], b, s);
            }
            sacc[t] = s;
        }

        // Online softmax per owned row (row = g*8 + j).
        float alpha[8];
        #pragma unroll
        for (int j = 0; j < 8; ++j) {
            float mx = fmaxf(fmaxf(sacc[0][j], sacc[1][j]),
                             fmaxf(sacc[2][j], sacc[3][j]));
            #pragma unroll
            for (int msk = 1; msk < 16; msk <<= 1)
                mx = fmaxf(mx, __shfl_xor(mx, msk, 32));
            float mn  = fmaxf(mrow[j], mx);
            alpha[j]  = __expf(mrow[j] - mn);
            mrow[j]   = mn;
            float rs = 0.0f;
            #pragma unroll
            for (int t = 0; t < 4; ++t) {
                float p = __expf(sacc[t][j] - mn);
                sacc[t][j] = p;
                rs += p;
            }
            #pragma unroll
            for (int msk = 1; msk < 16; msk <<= 1)
                rs += __shfl_xor(rs, msk, 32);
            lrow[j] = lrow[j] * alpha[j] + rs;
        }
        #pragma unroll
        for (int ot = 0; ot < 8; ++ot)
            #pragma unroll
            for (int j = 0; j < 8; ++j)
                oacc[ot][j] *= alpha[j];

        // P: C-layout -> row-major LDS (per-wave region; same-wave DS order).
        #pragma unroll
        for (int t = 0; t < 4; ++t)
            #pragma unroll
            for (int j = 0; j < 8; ++j)
                Ps[w][g * 8 + j][t * 16 + ln] = (_Float16)sacc[t][j];

        // P A-fragments: 16x64 -> two 16x32 chunks.
        v16h pa[2];
        #pragma unroll
        for (int c = 0; c < 2; ++c)
            #pragma unroll
            for (int j = 0; j < 8; ++j) {
                int kb = 32 * c + g * 16 + 2 * j;
                pa[c][2 * j]     = Ps[w][ln][kb];
                pa[c][2 * j + 1] = Ps[w][ln][kb + 1];
            }

        // O(16x128) += P(16x64) @ V(64x128); V consumed transposed.
        #pragma unroll
        for (int ot = 0; ot < 8; ++ot)
            #pragma unroll
            for (int c = 0; c < 2; ++c) {
                v16h b;
                #pragma unroll
                for (int j = 0; j < 8; ++j) {
                    int kk = 32 * c + g * 16 + 2 * j;
                    b[2 * j]     = Vst[ot * 16 + ln][kk];
                    b[2 * j + 1] = Vst[ot * 16 + ln][kk + 1];
                }
                oacc[ot] = wmma_f16(pa[c], b, oacc[ot]);
            }
        __syncthreads();
    }

    // Normalize and write (S, H, Dh) -> (S, D_MODEL) row-major.
    #pragma unroll
    for (int ot = 0; ot < 8; ++ot)
        #pragma unroll
        for (int j = 0; j < 8; ++j) {
            int row = q0 + w * 16 + g * 8 + j;
            int col = h * D_H + ot * 16 + ln;
            out[(size_t)row * D_MODEL + col] = oacc[ot][j] / lrow[j];
        }
}

// ---------------------------------------------------------------------------
// Launch: 6 projection GEMMs, RoPE, flash attention, output GEMM (+bias).
// ---------------------------------------------------------------------------
extern "C" void kernel_launch(void* const* d_in, const int* in_sizes, int n_in,
                              void* d_out, int out_size, void* d_ws, size_t ws_size,
                              hipStream_t stream)
{
    (void)in_sizes; (void)n_in; (void)out_size; (void)ws_size;

    const float* x     = (const float*)d_in[0];
    const float* W_DQ  = (const float*)d_in[1];
    const float* W_UQ  = (const float*)d_in[2];
    const float* W_DKV = (const float*)d_in[3];
    const float* W_UK  = (const float*)d_in[4];
    const float* W_UV  = (const float*)d_in[5];
    const float* W_Kr  = (const float*)d_in[6];
    const float* W_O   = (const float*)d_in[7];
    const float* b_O   = (const float*)d_in[8];
    const int*   past  = (const int*)d_in[9];
    float*       out   = (float*)d_out;

    float* ws = (float*)d_ws;
    size_t off = 0;
    float* c_kv   = ws + off; off += (size_t)SEQ * PROJ;     // 16.8 MB
    float* k_nope = ws + off; off += (size_t)SEQ * PROJ;     // 16.8 MB
    float* vbuf   = ws + off; off += (size_t)SEQ * D_MODEL;  // 33.6 MB
    float* qbuf   = ws + off; off += (size_t)SEQ * D_MODEL;  // 33.6 MB
    float* krope  = ws + off; off += (size_t)SEQ * QK_ROPE;  // 0.5 MB
    float* a_out  = ws + off; off += (size_t)SEQ * D_MODEL;  // 33.6 MB
    float* c_q    = a_out;  // live ranges don't overlap: reuse

    const dim3 blk(256);
    const int  MT = SEQ / 128;

    // c_q = x @ W_DQ                (2048x2048, K=4096)
    gemm_wmma_f16<false><<<dim3(PROJ / 128, MT), blk, 0, stream>>>(
        x, W_DQ, nullptr, c_q, SEQ, PROJ, D_MODEL);
    // q = c_q @ W_UQ                (2048x4096, K=2048)
    gemm_wmma_f16<false><<<dim3(D_MODEL / 128, MT), blk, 0, stream>>>(
        c_q, W_UQ, nullptr, qbuf, SEQ, D_MODEL, PROJ);
    // c_kv = x @ W_DKV              (2048x2048, K=4096)
    gemm_wmma_f16<false><<<dim3(PROJ / 128, MT), blk, 0, stream>>>(
        x, W_DKV, nullptr, c_kv, SEQ, PROJ, D_MODEL);
    // k_nope = c_kv @ W_UK          (2048x2048, K=2048)
    gemm_wmma_f16<false><<<dim3(PROJ / 128, MT), blk, 0, stream>>>(
        c_kv, W_UK, nullptr, k_nope, SEQ, PROJ, PROJ);
    // v = c_kv @ W_UV               (2048x4096, K=2048)
    gemm_wmma_f16<false><<<dim3(D_MODEL / 128, MT), blk, 0, stream>>>(
        c_kv, W_UV, nullptr, vbuf, SEQ, D_MODEL, PROJ);
    // k_rope = x @ W_K_rope         (2048x64, K=4096) — N-edge guarded tile
    gemm_wmma_f16<false><<<dim3((QK_ROPE + 127) / 128, MT), blk, 0, stream>>>(
        x, W_Kr, nullptr, krope, SEQ, QK_ROPE, D_MODEL);

    // RoPE in place on q rope-halves (32 heads) + k_rope (y == N_HEADS slot).
    rope_kernel<<<dim3(SEQ, N_HEADS + 1), 64, 0, stream>>>(qbuf, krope, past);

    // Flash attention.
    attn_flash_wmma<<<dim3(SEQ / 128, N_HEADS), blk, 0, stream>>>(
        qbuf, k_nope, krope, vbuf, a_out);

    // out = attn_out @ W_O + b_O    (2048x4096, K=4096)
    gemm_wmma_f16<true><<<dim3(D_MODEL / 128, MT), blk, 0, stream>>>(
        a_out, W_O, b_O, out, SEQ, D_MODEL, D_MODEL);
}
